// CausalSelfAttention_81509889343604
// MI455X (gfx1250) — compile-verified, matrix operand order checked by reading, not run on hardware
//
#include <hip/hip_runtime.h>
#include <hip/hip_bf16.h>

// ---------------- problem constants ----------------
constexpr int kB  = 2;
constexpr int kT  = 2048;
constexpr int kC  = 2048;
constexpr int kH  = 16;
constexpr int kD  = 128;          // head dim
constexpr int k3C = 3 * kC;      // 6144
constexpr float kScale = 0.08838834764831845f;  // 1/sqrt(128)

typedef __attribute__((ext_vector_type(16))) __bf16 v16bf;
typedef __attribute__((ext_vector_type(8)))  __bf16 v8bf;
typedef __attribute__((ext_vector_type(8)))  float  v8f;
typedef __attribute__((ext_vector_type(4)))  int    v4i;

// ---- async global->LDS path (gfx1250 ASYNCcnt pipe), guarded so we always compile
#if defined(__has_builtin)
#if __has_builtin(__builtin_amdgcn_global_load_async_to_lds_b128) && \
    __has_builtin(__builtin_amdgcn_s_wait_asynccnt)
#define USE_ASYNC_LDS 1
#endif
#endif
#ifndef USE_ASYNC_LDS
#define USE_ASYNC_LDS 0
#endif

#if USE_ASYNC_LDS
typedef __attribute__((address_space(1))) v4i* as1_v4i;
typedef __attribute__((address_space(3))) v4i* as3_v4i;
// Per the CDNA5 aperture rules, the low 32 bits of a generic LDS address are
// the LDS byte offset, and a generic global address equals the AS1 address.
__device__ __forceinline__ void async_copy_b128(const unsigned short* gsrc,
                                                unsigned short* ldst) {
    as1_v4i g = (as1_v4i)(unsigned long long)(__SIZE_TYPE__)gsrc;
    as3_v4i l = (as3_v4i)(unsigned)(__SIZE_TYPE__)ldst;
    __builtin_amdgcn_global_load_async_to_lds_b128(g, l, 0, 0);
}
#endif

// ---------------- bf16 helpers (bit-exact RNE, no cvt dependency) ----------------
__device__ __forceinline__ unsigned short f2bf(float f) {
    unsigned u = __builtin_bit_cast(unsigned, f);
    u += 0x7fffu + ((u >> 16) & 1u);
    return (unsigned short)(u >> 16);
}
__device__ __forceinline__ float bf2f(unsigned short h) {
    unsigned u = ((unsigned)h) << 16;
    return __builtin_bit_cast(float, u);
}

// Load a 16-element bf16 fragment for one lane from a K-contiguous row.
// Matches ISA 16-bit A/B layout:
//  elements 0..7  = K[half*8 .. half*8+7]         (one b128 load)
//  elements 8..15 = K[16+half*8 .. 16+half*8+7]   (one b128 load)
__device__ __forceinline__ v16bf load_frag(const unsigned short* rowp, int half) {
    v8bf lo = *reinterpret_cast<const v8bf*>(rowp + half * 8);
    v8bf hi = *reinterpret_cast<const v8bf*>(rowp + 16 + half * 8);
    return __builtin_shufflevector(lo, hi, 0,1,2,3,4,5,6,7,8,9,10,11,12,13,14,15);
}

__device__ __forceinline__ v8f wmma_bf16(v16bf a, v16bf b, v8f c) {
    return __builtin_amdgcn_wmma_f32_16x16x32_bf16(
        /*neg_a=*/false, a, /*neg_b=*/false, b,
        /*c_mod=*/(short)0, c, /*reuse_a=*/false, /*reuse_b=*/false);
}

// ---------------- kernel 1: f32 -> bf16 conversions + weight transposes ----------------
__global__ void prep_kernel(const float* __restrict__ x,
                            const float* __restrict__ wqkv,
                            const float* __restrict__ wout,
                            unsigned short* __restrict__ xb,
                            unsigned short* __restrict__ wqkvT,
                            unsigned short* __restrict__ woutT) {
    const size_t n1 = (size_t)kB * kT * kC;       // x
    const size_t n2 = (size_t)k3C * kC;           // wqkvT [3C][C]
    const size_t n3 = (size_t)kC * kC;            // woutT [C][C]
    const size_t total = n1 + n2 + n3;
    for (size_t i = (size_t)blockIdx.x * blockDim.x + threadIdx.x; i < total;
         i += (size_t)gridDim.x * blockDim.x) {
        if (i < n1) {
            xb[i] = f2bf(x[i]);
        } else if (i < n1 + n2) {
            size_t j = i - n1;
            size_t n = j / kC, k = j - n * kC;
            wqkvT[j] = f2bf(wqkv[k * (size_t)k3C + n]);   // w_qkv is [C, 3C]
        } else {
            size_t j = i - n1 - n2;
            size_t n = j / kC, k = j - n * kC;
            woutT[j] = f2bf(wout[k * (size_t)kC + n]);    // w_out is [C, C]
        }
    }
}

// ---------------- register-blocked 64x64 GEMM core (per wave) ----------------
// A: row-major [.., K], rows row0..row0+63  (K-contiguous)
// B: row-major transposed weights [N, K], rows col0..col0+63 (K-contiguous)
__device__ __forceinline__ void gemm_core_64x64(const unsigned short* __restrict__ A,
                                                const unsigned short* __restrict__ B,
                                                int row0, int col0, int K,
                                                int m, int half, v8f acc[4][4]) {
    #pragma unroll
    for (int mt = 0; mt < 4; ++mt)
        #pragma unroll
        for (int nt = 0; nt < 4; ++nt)
            acc[mt][nt] = (v8f){0.f,0.f,0.f,0.f,0.f,0.f,0.f,0.f};

    const unsigned short* arow[4];
    const unsigned short* brow[4];
    #pragma unroll
    for (int mt = 0; mt < 4; ++mt)
        arow[mt] = A + (size_t)(row0 + mt * 16 + m) * K;
    #pragma unroll
    for (int nt = 0; nt < 4; ++nt)
        brow[nt] = B + (size_t)(col0 + nt * 16 + m) * K;

    #pragma unroll 1
    for (int k0 = 0; k0 < K; k0 += 32) {
        v16bf a[4], b[4];
        #pragma unroll
        for (int mt = 0; mt < 4; ++mt) a[mt] = load_frag(arow[mt] + k0, half);
        #pragma unroll
        for (int nt = 0; nt < 4; ++nt) b[nt] = load_frag(brow[nt] + k0, half);
        #pragma unroll
        for (int mt = 0; mt < 4; ++mt)
            #pragma unroll
            for (int nt = 0; nt < 4; ++nt)
                acc[mt][nt] = wmma_bf16(a[mt], b[nt], acc[mt][nt]);
    }
}

// ---------------- kernel 2: QKV GEMM (bf16 WMMA, f32 accum) ----------------
// 4 waves/block; wave w computes a 64x64 tile at (row0, bx*256 + w*64).
// Epilogue scatters into q/k [B,H,T,D] and V-transposed [B,H,D,T].
__global__ __launch_bounds__(128)
void qkv_gemm_kernel(const unsigned short* __restrict__ xb,
                     const unsigned short* __restrict__ wqkvT,
                     unsigned short* __restrict__ qraw,
                     unsigned short* __restrict__ kraw,
                     unsigned short* __restrict__ vt) {
    const int lane = threadIdx.x & 31;
    const int wave = threadIdx.x >> 5;
    const int m    = lane & 15;
    const int half = lane >> 4;
    const int col0 = blockIdx.x * 256 + wave * 64;   // 0..6143
    const int row0 = blockIdx.y * 64;                // 0..B*T-1

    v8f acc[4][4];
    gemm_core_64x64(xb, wqkvT, row0, col0, kC, m, half, acc);

    #pragma unroll
    for (int nt = 0; nt < 4; ++nt) {
        const int j   = col0 + nt * 16 + m;    // output column
        const int sec = j / kC;                // 0=q 1=k 2=v
        const int f   = j - sec * kC;
        const int hh  = f / kD;
        const int dd  = f - hh * kD;
        #pragma unroll
        for (int mt = 0; mt < 4; ++mt) {
            #pragma unroll
            for (int v = 0; v < 8; ++v) {
                const int r  = row0 + mt * 16 + v + 8 * half;
                const int b_ = r / kT;
                const int t  = r - b_ * kT;
                const unsigned short val = f2bf(acc[mt][nt][v]);
                if (sec == 2) {
                    vt[(((size_t)(b_ * kH + hh)) * kD + dd) * kT + t] = val;
                } else {
                    const size_t idx = (((size_t)(b_ * kH + hh)) * kT + t) * kD + dd;
                    if (sec == 0) qraw[idx] = val; else kraw[idx] = val;
                }
            }
        }
    }
}

// ---------------- kernel 3: RoPE on q,k (q pre-scaled by 1/sqrt(D)) ----------------
__global__ void rope_kernel(const unsigned short* __restrict__ qraw,
                            const unsigned short* __restrict__ kraw,
                            const float* __restrict__ cosT,
                            const float* __restrict__ sinT,
                            unsigned short* __restrict__ qr,
                            unsigned short* __restrict__ kr) {
    const int bht = blockIdx.x;           // (b*H + h)*T + t
    const int d   = threadIdx.x;          // 0..127
    const int t   = bht % kT;
    const size_t base = (size_t)bht * kD;
    const float c = cosT[(size_t)t * kD + d];
    const float s = sinT[(size_t)t * kD + d];
    const int   dp = (d < kD / 2) ? d + kD / 2 : d - kD / 2;
    const float sgn = (d < kD / 2) ? -1.f : 1.f;

    const float q1 = bf2f(qraw[base + d]);
    const float q2 = bf2f(qraw[base + dp]);
    const float k1 = bf2f(kraw[base + d]);
    const float k2 = bf2f(kraw[base + dp]);
    qr[base + d] = f2bf((q1 * c + sgn * q2 * s) * kScale);
    kr[base + d] = f2bf(k1 * c + sgn * k2 * s);
}

// ---------------- kernel 4: causal flash attention ----------------
// 4 waves/block cover a 64-row q band; each 32-key K/V chunk is staged once
// into LDS (async global->LDS when available) and consumed by all 4 waves.
// kv loop bound is block-uniform; fully-masked chunks for early waves
// contribute exactly zero through the online softmax.
__global__ __launch_bounds__(128)
void flash_attn_kernel(const unsigned short* __restrict__ qr,
                       const unsigned short* __restrict__ kr,
                       const unsigned short* __restrict__ vt,
                       unsigned short* __restrict__ attn) {
    const int tid   = threadIdx.x;
    const int lane  = tid & 31;
    const int wave  = tid >> 5;
    const int m     = lane & 15;
    const int half  = lane >> 4;
    const int tq0   = blockIdx.x * 64;     // block q base
    const int bh    = blockIdx.y;          // 0..B*H-1
    const int b_    = bh / kH;
    const int h_    = bh % kH;
    const int t0    = tq0 + wave * 16;     // this wave's q rows

    const unsigned short* qbase = qr + ((size_t)bh * kT + t0) * kD;
    const unsigned short* kbase = kr + (size_t)bh * kT * kD;
    const unsigned short* vbase = vt + (size_t)bh * kD * kT;

    // K tile: 32 keys x 128 d   (row stride 136 el -> 272B, 16B aligned)
    // V tile: 128 d x 32 keys   (row stride 40 el  -> 80B,  16B aligned)
    // P tile: per-wave 16 x 32  (row stride 40 el)
    __shared__ unsigned short kl[32 * 136];
    __shared__ unsigned short vl[128 * 40];
    __shared__ unsigned short psh[4 * 16 * 40];
    unsigned short* pw = &psh[wave * 16 * 40];

    // Q fragments, whole D=128 kept in registers (4 chunks of K=32)
    v16bf qf[4];
    #pragma unroll
    for (int kc = 0; kc < 4; ++kc)
        qf[kc] = load_frag(qbase + (size_t)m * kD + kc * 32, half);

    v8f o[8];
    #pragma unroll
    for (int nd = 0; nd < 8; ++nd)
        o[nd] = (v8f){0.f,0.f,0.f,0.f,0.f,0.f,0.f,0.f};
    float mrow[8], lrow[8];
    #pragma unroll
    for (int v = 0; v < 8; ++v) { mrow[v] = -1e30f; lrow[v] = 0.f; }

    const int nk_max = tq0 + 64;           // keys needed by the last wave
    for (int kv0 = 0; kv0 < nk_max; kv0 += 32) {
        __syncthreads();   // WAR: previous iteration's LDS reads done
        // ---- cooperative stage of K chunk (32 x 128) ----
        {
            const int row = tid >> 2;             // 0..31
            const int seg = (tid & 3) * 32;       // 0,32,64,96
            const unsigned short* src = kbase + (size_t)(kv0 + row) * kD + seg;
            unsigned short* dst = &kl[row * 136 + seg];
            #pragma unroll
            for (int q = 0; q < 4; ++q) {
#if USE_ASYNC_LDS
                async_copy_b128(src + q * 8, dst + q * 8);
#else
                *reinterpret_cast<v8bf*>(dst + q * 8) =
                    *reinterpret_cast<const v8bf*>(src + q * 8);
#endif
            }
        }
        // ---- cooperative stage of V chunk (128 x 32) ----
        {
            const unsigned short* src = vbase + (size_t)tid * kT + kv0;
            unsigned short* dst = &vl[tid * 40];
            #pragma unroll
            for (int q = 0; q < 4; ++q) {
#if USE_ASYNC_LDS
                async_copy_b128(src + q * 8, dst + q * 8);
#else
                *reinterpret_cast<v8bf*>(dst + q * 8) =
                    *reinterpret_cast<const v8bf*>(src + q * 8);
#endif
            }
        }
#if USE_ASYNC_LDS
        __builtin_amdgcn_s_wait_asynccnt(0);   // our async copies landed in LDS
#endif
        __syncthreads();                        // all waves' copies visible

        // ---- scores: two 16x16 tiles over the 32 staged keys ----
        v8f s0 = (v8f){0.f,0.f,0.f,0.f,0.f,0.f,0.f,0.f};
        v8f s1 = (v8f){0.f,0.f,0.f,0.f,0.f,0.f,0.f,0.f};
        #pragma unroll
        for (int kc = 0; kc < 4; ++kc) {
            s0 = wmma_bf16(qf[kc], load_frag(&kl[(m)      * 136 + kc * 32], half), s0);
            s1 = wmma_bf16(qf[kc], load_frag(&kl[(16 + m) * 136 + kc * 32], half), s1);
        }
        // causal mask: key index > query index -> -inf
        #pragma unroll
        for (int v = 0; v < 8; ++v) {
            const int tq = t0 + v + 8 * half;
            if (kv0 + m      > tq) s0[v] = -1e30f;
            if (kv0 + 16 + m > tq) s1[v] = -1e30f;
        }
        // ---- online softmax stats (row reductions across 16-lane groups) ----
        float scv[8];
        #pragma unroll
        for (int v = 0; v < 8; ++v) {
            float mx = fmaxf(s0[v], s1[v]);
            #pragma unroll
            for (int off = 8; off; off >>= 1) mx = fmaxf(mx, __shfl_xor(mx, off));
            const float mnew = fmaxf(mrow[v], mx);
            const float sc   = __expf(mrow[v] - mnew);
            const float p0   = __expf(s0[v] - mnew);
            const float p1   = __expf(s1[v] - mnew);
            float rs = p0 + p1;
            #pragma unroll
            for (int off = 8; off; off >>= 1) rs += __shfl_xor(rs, off);
            lrow[v] = lrow[v] * sc + rs;
            mrow[v] = mnew;
            scv[v]  = sc;
            s0[v] = p0; s1[v] = p1;
        }
        #pragma unroll
        for (int nd = 0; nd < 8; ++nd)
            #pragma unroll
            for (int v = 0; v < 8; ++v) o[nd][v] *= scv[v];

        // ---- P: C-layout accumulator -> A-fragment layout via per-wave LDS ----
        #pragma unroll
        for (int v = 0; v < 8; ++v) {
            const int rr = v + 8 * half;
            pw[rr * 40 + m]      = f2bf(s0[v]);
            pw[rr * 40 + 16 + m] = f2bf(s1[v]);
        }
        __syncthreads();
        const v16bf pf = load_frag(&pw[m * 40], half);

        // ---- O += P @ V (8 d-tiles, K=32 keys, V fragments from LDS) ----
        #pragma unroll
        for (int nd = 0; nd < 8; ++nd)
            o[nd] = wmma_bf16(pf, load_frag(&vl[(nd * 16 + m) * 40], half), o[nd]);
    }

    // ---- finalize and store bf16 attention output [B, T, H*D] ----
    float inv[8];
    #pragma unroll
    for (int v = 0; v < 8; ++v) inv[v] = 1.f / lrow[v];
    #pragma unroll
    for (int nd = 0; nd < 8; ++nd)
        #pragma unroll
        for (int v = 0; v < 8; ++v) {
            const int t = t0 + v + 8 * half;
            attn[((size_t)b_ * kT + t) * kC + h_ * kD + nd * 16 + m] =
                f2bf(o[nd][v] * inv[v]);
        }
}

// ---------------- kernel 5: output projection GEMM -> f32 d_out ----------------
__global__ __launch_bounds__(128)
void proj_gemm_kernel(const unsigned short* __restrict__ attn,
                      const unsigned short* __restrict__ woutT,
                      float* __restrict__ out) {
    const int lane = threadIdx.x & 31;
    const int wave = threadIdx.x >> 5;
    const int m    = lane & 15;
    const int half = lane >> 4;
    const int col0 = blockIdx.x * 256 + wave * 64;
    const int row0 = blockIdx.y * 64;

    v8f acc[4][4];
    gemm_core_64x64(attn, woutT, row0, col0, kC, m, half, acc);

    #pragma unroll
    for (int mt = 0; mt < 4; ++mt)
        #pragma unroll
        for (int nt = 0; nt < 4; ++nt)
            #pragma unroll
            for (int v = 0; v < 8; ++v)
                out[(size_t)(row0 + mt * 16 + v + 8 * half) * kC + col0 + nt * 16 + m] =
                    acc[mt][nt][v];
}

// ---------------- host side ----------------
extern "C" void kernel_launch(void* const* d_in, const int* in_sizes, int n_in,
                              void* d_out, int out_size, void* d_ws, size_t ws_size,
                              hipStream_t stream) {
    const float* x    = (const float*)d_in[0];
    const float* cosT = (const float*)d_in[1];
    const float* sinT = (const float*)d_in[2];
    const float* wqkv = (const float*)d_in[3];
    const float* wout = (const float*)d_in[4];
    float* out = (float*)d_out;

    unsigned short* ws = (unsigned short*)d_ws;
    const size_t nBTC = (size_t)kB * kT * kC;          // 8,388,608
    unsigned short* xb     = ws;
    unsigned short* wqkvT  = xb    + nBTC;
    unsigned short* woutT  = wqkvT + (size_t)k3C * kC;
    unsigned short* qraw   = woutT + (size_t)kC * kC;
    unsigned short* kraw   = qraw  + nBTC;
    unsigned short* vt     = kraw  + nBTC;
    unsigned short* qrot   = vt    + nBTC;
    unsigned short* krot   = qrot  + nBTC;
    unsigned short* attn   = qraw;                     // alias: qraw dead after RoPE

    // 1) precision prep + weight transposes
    prep_kernel<<<4096, 256, 0, stream>>>(x, wqkv, wout, xb, wqkvT, woutT);

    // 2) QKV GEMM (bf16 WMMA, 64x64 per wave)
    {
        dim3 grid(k3C / 256, (kB * kT) / 64);
        qkv_gemm_kernel<<<grid, 128, 0, stream>>>(xb, wqkvT, qraw, kraw, vt);
    }

    // 3) RoPE
    rope_kernel<<<kB * kH * kT, kD, 0, stream>>>(qraw, kraw, cosT, sinT, qrot, krot);

    // 4) causal flash attention (64-row q band per block, K/V shared via LDS)
    {
        dim3 grid(kT / 64, kB * kH);
        flash_attn_kernel<<<grid, 128, 0, stream>>>(qrot, krot, vt, attn);
    }

    // 5) output projection
    {
        dim3 grid(kC / 256, (kB * kT) / 64);
        proj_gemm_kernel<<<grid, 128, 0, stream>>>(attn, woutT, out);
    }
}
